// QWQBlockWithLoRA_60284160967081
// MI455X (gfx1250) — matrix operand order repaired and checked by reading, not compiled, not run on hardware
//
#include <hip/hip_runtime.h>
#include <hip/hip_bf16.h>

typedef __attribute__((ext_vector_type(16))) _Float16 v16h;
typedef __attribute__((ext_vector_type(8)))  _Float16 h8;
typedef __attribute__((ext_vector_type(8)))  float    v8f;
typedef int i4v __attribute__((vector_size(4 * sizeof(int))));

// ---------------------------------------------------------------------------
// Async global->LDS staging if the toolchain exposes it (gfx1250 ASYNCcnt
// path); otherwise fall back to global_load -> ds_store staging.
// Probe result (round 3): builtin EXISTS, 4 args, arg0 = global int4*,
// prints as '__device__' (target AS1). s_wait_asynccnt(int) also exists.
// ---------------------------------------------------------------------------
#if defined(__AMDGCN__) && \
    __has_builtin(__builtin_amdgcn_global_load_async_to_lds_b128) && \
    __has_builtin(__builtin_amdgcn_s_wait_asynccnt)
#define USE_ASYNC_LDS 1
#else
#define USE_ASYNC_LDS 0
#endif

#if USE_ASYNC_LDS
typedef __attribute__((address_space(1))) i4v* gas_i4p;
typedef __attribute__((address_space(3))) i4v* las_i4p;

__device__ __forceinline__ void async_cp16(const _Float16* g, _Float16* l) {
    __builtin_amdgcn_global_load_async_to_lds_b128(
        (gas_i4p)(uintptr_t)g,
        (las_i4p)(unsigned)(uintptr_t)l,
        0, 0);
}
#endif

// ---------------------------------------------------------------------------
// LDS fragment loaders for v_wmma_f32_16x16x32_f16 (wave32).
//   A tile in LDS: [128 rows][40 halves] (32 data + 8 pad)
//   B tile in LDS: [32 rows][136 halves] (128 data + 8 pad)
// A frag: lane = M row (lane&15), lh = lane>>4:
//   elements 0..7 = K 8*lh..+7 ; elements 8..15 = K 16+8*lh..+7
// B frag: lane = N column (lane&15), lh = lane>>4:
//   elements 0..7 = K 8*lh..+7 ; elements 8..15 = K 16+8*lh..+7
// ---------------------------------------------------------------------------
__device__ __forceinline__ v16h lds_a_frag(const _Float16* base, int lane) {
    const int m  = lane & 15;
    const int lh = lane >> 4;
    const _Float16* p = base + m * 40 + 8 * lh;
    const h8 lo = *reinterpret_cast<const h8*>(p);
    const h8 hi = *reinterpret_cast<const h8*>(p + 16);
    return __builtin_shufflevector(lo, hi, 0,1,2,3,4,5,6,7,8,9,10,11,12,13,14,15);
}

__device__ __forceinline__ v16h lds_b_frag(const _Float16* base, int lane) {
    const int n  = lane & 15;
    const int lh = lane >> 4;
    v16h r;
#pragma unroll
    for (int s = 0; s < 16; ++s) {
        const int k = (s < 8) ? (8 * lh + s) : (16 + 8 * lh + (s - 8));
        r[s] = base[k * 136 + n];
    }
    return r;
}

// ---------------------------------------------------------------------------
// GEMM: C[M,Nout] = alpha * (A[M,K] @ B[K,Nout])   (optional ReLU)
//   CMODE 0: C f32 row-major ; 1: C f16 row-major ; 2: C f16 transposed
// Workgroup tile 128x128, 8 waves (4M x 2N), wave tile 32x64, K-step 32.
// Double-buffered LDS staging; every tile element hits LDS exactly once.
// ---------------------------------------------------------------------------
template <int CMODE, bool RELU>
__global__ __launch_bounds__(256)
void gemm_wmma(const _Float16* __restrict__ A, const _Float16* __restrict__ B,
               void* __restrict__ Cv, int M, int Nout, int K, float alpha)
{
    __shared__ _Float16 As[2][128 * 40];
    __shared__ _Float16 Bs[2][32 * 136];

    const int tid  = threadIdx.x;
    const int lane = tid & 31;
    const int wave = tid >> 5;
    const int wm   = wave >> 1;                 // 0..3
    const int wn   = wave & 1;                  // 0..1
    const int mBlock = blockIdx.y * 128;
    const int nBlock = blockIdx.x * 128;
    const int mBase  = mBlock + wm * 32;
    const int nBase  = nBlock + wn * 64;

    // Cooperative staging: 256 threads x 2 x 16B chunks per tile.
    const _Float16* gA = A + (size_t)(mBlock + (tid >> 2)) * K + (tid & 3) * 8;
    const _Float16* gB = B + (size_t)(tid >> 4) * Nout + nBlock + (tid & 15) * 8;
    const size_t gA2 = (size_t)64 * K;          // second A chunk (+64 rows)
    const size_t gB2 = (size_t)16 * Nout;       // second B chunk (+16 rows)
    const int lA = (tid >> 2) * 40 + (tid & 3) * 8;
    const int lB = (tid >> 4) * 136 + (tid & 15) * 8;

    v8f acc[2][4];
#pragma unroll
    for (int i = 0; i < 2; ++i)
#pragma unroll
        for (int j = 0; j < 4; ++j)
            acc[i][j] = (v8f){0.f,0.f,0.f,0.f,0.f,0.f,0.f,0.f};

    const int nsteps = K >> 5;

#if USE_ASYNC_LDS
    async_cp16(gA,       &As[0][lA]);
    async_cp16(gA + gA2, &As[0][lA + 64 * 40]);
    async_cp16(gB,       &Bs[0][lB]);
    async_cp16(gB + gB2, &Bs[0][lB + 16 * 136]);
    __builtin_amdgcn_s_wait_asynccnt(0);
    __syncthreads();
#else
    {
        uint4 ra0 = *reinterpret_cast<const uint4*>(gA);
        uint4 ra1 = *reinterpret_cast<const uint4*>(gA + gA2);
        uint4 rb0 = *reinterpret_cast<const uint4*>(gB);
        uint4 rb1 = *reinterpret_cast<const uint4*>(gB + gB2);
        *reinterpret_cast<uint4*>(&As[0][lA])            = ra0;
        *reinterpret_cast<uint4*>(&As[0][lA + 64 * 40])  = ra1;
        *reinterpret_cast<uint4*>(&Bs[0][lB])            = rb0;
        *reinterpret_cast<uint4*>(&Bs[0][lB + 16 * 136]) = rb1;
        __syncthreads();
    }
#endif

    for (int s = 0; s < nsteps; ++s) {
        const int cur  = s & 1;
        const bool more = (s + 1 < nsteps);

#if USE_ASYNC_LDS
        if (more) {
            gA += 32;
            gB += (size_t)32 * Nout;
            async_cp16(gA,       &As[cur ^ 1][lA]);
            async_cp16(gA + gA2, &As[cur ^ 1][lA + 64 * 40]);
            async_cp16(gB,       &Bs[cur ^ 1][lB]);
            async_cp16(gB + gB2, &Bs[cur ^ 1][lB + 16 * 136]);
            if (s + 2 < nsteps) {
                __builtin_prefetch(gA + 32, 0, 1);
                __builtin_prefetch(gB + (size_t)32 * Nout, 0, 1);
            }
        }
#else
        uint4 na0, na1, nb0, nb1;
        if (more) {
            gA += 32;
            gB += (size_t)32 * Nout;
            na0 = *reinterpret_cast<const uint4*>(gA);
            na1 = *reinterpret_cast<const uint4*>(gA + gA2);
            nb0 = *reinterpret_cast<const uint4*>(gB);
            nb1 = *reinterpret_cast<const uint4*>(gB + gB2);
            if (s + 2 < nsteps) {
                __builtin_prefetch(gA + 32, 0, 1);
                __builtin_prefetch(gB + (size_t)32 * Nout, 0, 1);
            }
        }
#endif

        // ---- compute on buffer `cur` ----
        const _Float16* Ab = &As[cur][wm * 32 * 40];
        const _Float16* Bb = &Bs[cur][wn * 64];
        const v16h a0 = lds_a_frag(Ab, lane);
        const v16h a1 = lds_a_frag(Ab + 16 * 40, lane);
        v16h b[4];
#pragma unroll
        for (int j = 0; j < 4; ++j)
            b[j] = lds_b_frag(Bb + j * 16, lane);
#pragma unroll
        for (int j = 0; j < 4; ++j) {
            acc[0][j] = __builtin_amdgcn_wmma_f32_16x16x32_f16(
                false, a0, false, b[j], (short)0, acc[0][j], false, false);
            acc[1][j] = __builtin_amdgcn_wmma_f32_16x16x32_f16(
                false, a1, false, b[j], (short)0, acc[1][j], false, false);
        }

        if (more) {
#if USE_ASYNC_LDS
            __builtin_amdgcn_s_wait_asynccnt(0);
            __syncthreads();
#else
            *reinterpret_cast<uint4*>(&As[cur ^ 1][lA])            = na0;
            *reinterpret_cast<uint4*>(&As[cur ^ 1][lA + 64 * 40])  = na1;
            *reinterpret_cast<uint4*>(&Bs[cur ^ 1][lB])            = nb0;
            *reinterpret_cast<uint4*>(&Bs[cur ^ 1][lB + 16 * 136]) = nb1;
            __syncthreads();
#endif
        }
    }

    const int nn = lane & 15;
    const int lh = lane >> 4;
#pragma unroll
    for (int i = 0; i < 2; ++i) {
#pragma unroll
        for (int j = 0; j < 4; ++j) {
            if (CMODE == 0) {
                float* C = (float*)Cv;
                float* cp = C + (size_t)(mBase + i * 16 + lh * 8) * Nout
                              + (nBase + j * 16 + nn);
#pragma unroll
                for (int r = 0; r < 8; ++r) {
                    float val = acc[i][j][r] * alpha;
                    if (RELU) val = fmaxf(val, 0.0f);
                    cp[(size_t)r * Nout] = val;
                }
            } else if (CMODE == 1) {
                _Float16* C = (_Float16*)Cv;
                _Float16* cp = C + (size_t)(mBase + i * 16 + lh * 8) * Nout
                                 + (nBase + j * 16 + nn);
#pragma unroll
                for (int r = 0; r < 8; ++r) {
                    float val = acc[i][j][r] * alpha;
                    if (RELU) val = fmaxf(val, 0.0f);
                    cp[(size_t)r * Nout] = (_Float16)val;
                }
            } else {
                _Float16* C = (_Float16*)Cv;
                h8 o;
#pragma unroll
                for (int r = 0; r < 8; ++r) {
                    float val = acc[i][j][r] * alpha;
                    if (RELU) val = fmaxf(val, 0.0f);
                    o[r] = (_Float16)val;
                }
                _Float16* cp = C + (size_t)(nBase + j * 16 + nn) * M
                                 + (mBase + i * 16 + lh * 8);
                *reinterpret_cast<h8*>(cp) = o;
            }
        }
    }
}

// ---------------------------------------------------------------------------
// W_eff^T = (W + s * B @ A)^T, stored f16 [K, Dout] row-major.
// ---------------------------------------------------------------------------
__global__ __launch_bounds__(256)
void make_weff_t(const float* __restrict__ W, const float* __restrict__ A,
                 const float* __restrict__ Bm, _Float16* __restrict__ Wt,
                 int Dout, int K, float s)
{
    size_t i = (size_t)blockIdx.x * blockDim.x + threadIdx.x;
    size_t total = (size_t)Dout * K;
    if (i >= total) return;
    int j = (int)(i % Dout);
    int k = (int)(i / Dout);
    float acc = W[(size_t)j * K + k];
#pragma unroll
    for (int r = 0; r < 4; ++r)
        acc += s * Bm[j * 4 + r] * A[(size_t)r * K + k];
    Wt[i] = (_Float16)acc;
}

// ---------------------------------------------------------------------------
// f32 -> f16 elementwise copy.
// ---------------------------------------------------------------------------
__global__ __launch_bounds__(256)
void f32_to_f16(const float* __restrict__ in, _Float16* __restrict__ out, size_t n)
{
    size_t i = (size_t)blockIdx.x * blockDim.x + threadIdx.x;
    if (i < n) out[i] = (_Float16)in[i];
}

// ---------------------------------------------------------------------------
// Row softmax, f32 in -> f16 out (output may alias input: exponentials are
// kept in registers; the source row is not re-read after the sum barrier).
// ---------------------------------------------------------------------------
__global__ __launch_bounds__(256)
void softmax_rows_h(const float* __restrict__ S, _Float16* __restrict__ P, int Ncols)
{
    __shared__ float red[256];
    const int tid = threadIdx.x;
    const float* p = S + (size_t)blockIdx.x * Ncols;
    _Float16* ph = P + (size_t)blockIdx.x * Ncols;

    float mx = -3.4e38f;
    float e[16];                                   // Ncols / 256
    for (int i = tid; i < Ncols; i += 256) mx = fmaxf(mx, p[i]);
    red[tid] = mx; __syncthreads();
    for (int s = 128; s > 0; s >>= 1) {
        if (tid < s) red[tid] = fmaxf(red[tid], red[tid + s]);
        __syncthreads();
    }
    mx = red[0]; __syncthreads();

    float sum = 0.f;
    int j = 0;
    for (int i = tid; i < Ncols; i += 256, ++j) {
        float ev = __expf(p[i] - mx);
        e[j] = ev;
        sum += ev;
    }
    red[tid] = sum; __syncthreads();
    for (int s = 128; s > 0; s >>= 1) {
        if (tid < s) red[tid] += red[tid + s];
        __syncthreads();
    }
    const float inv = 1.0f / red[0];

    j = 0;
    for (int i = tid; i < Ncols; i += 256, ++j)
        ph[i] = (_Float16)(e[j] * inv);
}

// ---------------------------------------------------------------------------
// out = LayerNorm(a + b) * g + bias ; optional extra f16 copy of the output.
// ---------------------------------------------------------------------------
template <bool DUAL>
__global__ __launch_bounds__(256)
void layernorm_add(const float* __restrict__ a, const float* __restrict__ b,
                   const float* __restrict__ g, const float* __restrict__ bias,
                   float* __restrict__ out, _Float16* __restrict__ outh, int D)
{
    __shared__ float s1[256];
    __shared__ float s2[256];
    const int tid = threadIdx.x;
    const size_t row = (size_t)blockIdx.x * D;

    float vals[4];
    float sum = 0.f, sq = 0.f;
    int j = 0;
    for (int i = tid; i < D; i += 256, ++j) {
        float v = a[row + i] + b[row + i];
        vals[j] = v;
        sum += v; sq += v * v;
    }
    s1[tid] = sum; s2[tid] = sq; __syncthreads();
    for (int s = 128; s > 0; s >>= 1) {
        if (tid < s) { s1[tid] += s1[tid + s]; s2[tid] += s2[tid + s]; }
        __syncthreads();
    }
    const float mean = s1[0] / (float)D;
    const float var  = s2[0] / (float)D - mean * mean;
    const float rs   = rsqrtf(var + 1e-5f);

    j = 0;
    for (int i = tid; i < D; i += 256, ++j) {
        float y = (vals[j] - mean) * rs * g[i] + bias[i];
        out[row + i] = y;
        if (DUAL) outh[row + i] = (_Float16)y;
    }
}

// ---------------------------------------------------------------------------
// Host-side orchestration.
// ---------------------------------------------------------------------------
extern "C" void kernel_launch(void* const* d_in, const int* in_sizes, int n_in,
                              void* d_out, int out_size, void* d_ws, size_t ws_size,
                              hipStream_t stream)
{
    (void)in_sizes; (void)n_in; (void)out_size; (void)ws_size;
    const int N = 4096, D = 1024, D2 = 2048;
    const float scaling = 16.0f / 4.0f;           // alpha / rank
    const float inv_sqrt_d = 0.03125f;            // 1/sqrt(1024)

    const float* q  = (const float*)d_in[0];
    const float* k  = (const float*)d_in[1];
    const float* v  = (const float*)d_in[2];
    const float* Wq = (const float*)d_in[3];
    const float* Aq = (const float*)d_in[4];
    const float* Bq = (const float*)d_in[5];
    const float* Wk = (const float*)d_in[6];
    const float* Ak = (const float*)d_in[7];
    const float* Bk = (const float*)d_in[8];
    const float* Wv = (const float*)d_in[9];
    const float* Av = (const float*)d_in[10];
    const float* Bv = (const float*)d_in[11];
    const float* W1 = (const float*)d_in[12];
    const float* A1 = (const float*)d_in[13];
    const float* B1 = (const float*)d_in[14];
    const float* W2 = (const float*)d_in[15];
    const float* A2 = (const float*)d_in[16];
    const float* B2 = (const float*)d_in[17];
    const float* g1 = (const float*)d_in[18];
    const float* b1 = (const float*)d_in[19];
    const float* g2 = (const float*)d_in[20];
    const float* b2 = (const float*)d_in[21];

    float* ws = (float*)d_ws;
    const size_t NM = (size_t)N * D;              // 4M elements
    const size_t DD = (size_t)D * D;              // 1M elements

    // f32 region
    float* scores = ws;                            // [N,N]   (4*NM f32)
    float* attn   = ws + 4 * NM;                   // [N,D]   (later: ffn)
    float* x      = ws + 5 * NM;                   // [N,D]
    // f16 region
    _Float16* H   = (_Float16*)(ws + 6 * NM);
    _Float16* qh    = H;                           // [N,D]   (later: xh)
    _Float16* kh    = H + 1 * NM;                  // [N,D]   (later: hh low)
    _Float16* vh    = H + 2 * NM;                  // [N,D]   (later: hh high)
    _Float16* qph   = H + 3 * NM;                  // [N,D]
    _Float16* kphT  = H + 4 * NM;                  // [D,N]
    _Float16* vph   = H + 5 * NM;                  // [N,D]
    _Float16* WqT   = H + 6 * NM;                  // [D,D]
    _Float16* WkT   = WqT + DD;                    // [D,D]
    _Float16* WvT   = WkT + DD;                    // [D,D]
    _Float16* W1T   = WvT + DD;                    // [D,2D]
    _Float16* W2T   = W1T + 2 * DD;                // [2D,D]
    _Float16* probsH = (_Float16*)scores;          // alias (safe, see softmax)
    _Float16* xh    = qh;                          // reuse
    _Float16* hh    = kh;                          // reuse kh+vh: [N,2D]
    float* ffn      = attn;                        // reuse

    const dim3 blk(256);

    // 1) f16 copies of the activations feeding the projection GEMMs
    f32_to_f16<<<(int)(NM / 256), blk, 0, stream>>>(q, qh, NM);
    f32_to_f16<<<(int)(NM / 256), blk, 0, stream>>>(k, kh, NM);
    f32_to_f16<<<(int)(NM / 256), blk, 0, stream>>>(v, vh, NM);

    // 2) Fold LoRA into transposed f16 effective weights: W_eff^T [K, Dout]
    make_weff_t<<<(int)((DD     + 255) / 256), blk, 0, stream>>>(Wq, Aq, Bq, WqT, D,  D,  scaling);
    make_weff_t<<<(int)((DD     + 255) / 256), blk, 0, stream>>>(Wk, Ak, Bk, WkT, D,  D,  scaling);
    make_weff_t<<<(int)((DD     + 255) / 256), blk, 0, stream>>>(Wv, Av, Bv, WvT, D,  D,  scaling);
    make_weff_t<<<(int)((2 * DD + 255) / 256), blk, 0, stream>>>(W1, A1, B1, W1T, D2, D,  scaling);
    make_weff_t<<<(int)((2 * DD + 255) / 256), blk, 0, stream>>>(W2, A2, B2, W2T, D,  D2, scaling);

    // 3) Projections (all-f16 GEMMs): qph/vph row-major, kphT transposed
    gemm_wmma<1, false><<<dim3(D / 128, N / 128), blk, 0, stream>>>(qh, WqT, qph,  N, D, D, 1.0f);
    gemm_wmma<2, false><<<dim3(D / 128, N / 128), blk, 0, stream>>>(kh, WkT, kphT, N, D, D, 1.0f);
    gemm_wmma<1, false><<<dim3(D / 128, N / 128), blk, 0, stream>>>(vh, WvT, vph,  N, D, D, 1.0f);

    // 4) scores = (qp @ kp^T)/sqrt(D) via B = kphT [D, N]; softmax -> f16 probs
    gemm_wmma<0, false><<<dim3(N / 128, N / 128), blk, 0, stream>>>(qph, kphT, scores, N, N, D, inv_sqrt_d);
    softmax_rows_h<<<N, blk, 0, stream>>>(scores, probsH, N);

    // 5) attn = probs @ vp  (A = probsH [N,N], B = vph [N, D])
    gemm_wmma<0, false><<<dim3(D / 128, N / 128), blk, 0, stream>>>(probsH, vph, attn, N, D, N, 1.0f);

    // 6) x = LN(q + attn) (f32 residual copy + f16 GEMM feed)
    layernorm_add<true><<<N, blk, 0, stream>>>(q, attn, g1, b1, x, xh, D);

    // 7) FFN: hh = relu(x @ W1^T) f16 ; ffn = hh @ W2^T f32
    gemm_wmma<1, true ><<<dim3(D2 / 128, N / 128), blk, 0, stream>>>(xh, W1T, hh,  N, D2, D,  1.0f);
    gemm_wmma<0, false><<<dim3(D / 128,  N / 128), blk, 0, stream>>>(hh, W2T, ffn, N, D,  D2, 1.0f);

    // 8) out = LN(ffn + x)
    layernorm_add<false><<<N, blk, 0, stream>>>(ffn, x, g2, b2, (float*)d_out, nullptr, D);
}
